// Attention_61830349193262
// MI455X (gfx1250) — matrix operand order repaired and checked by reading, not compiled, not run on hardware
//
#include <hip/hip_runtime.h>
#include <math.h>

// ---------------------------------------------------------------------------
// Feature detection (probe-verified on this toolchain, round 4):
//   tdm_header=YES  -> 6-arg __builtin_amdgcn_tensor_load_to_lds
//   global_load_async_to_lds_b128=YES, s_wait_asynccnt=YES, s_wait_tensorcnt=YES
// Round-5 diagnostic: async builtin takes (v4i AS1*, v4i AS3*, imm, imm).
// ---------------------------------------------------------------------------
#if defined(__has_include)
#  if __has_include(<hip/amd_detail/amd_gfx1250_TDM.h>)
#    define TDM_HEADER_TOOLCHAIN 1
#  else
#    define TDM_HEADER_TOOLCHAIN 0
#  endif
#else
#  define TDM_HEADER_TOOLCHAIN 0
#endif

#if __has_builtin(__builtin_amdgcn_tensor_load_to_lds) && \
    __has_builtin(__builtin_amdgcn_s_wait_tensorcnt)
#  define HAVE_TDM 1
#else
#  define HAVE_TDM 0
#endif

#if __has_builtin(__builtin_amdgcn_global_load_async_to_lds_b128) && \
    __has_builtin(__builtin_amdgcn_s_wait_asynccnt)
#  define HAVE_ASYNC_LDS 1
#else
#  define HAVE_ASYNC_LDS 0
#endif

// Problem dims (fixed by the reference)
#define B_DIM 8
#define S_DIM 2048   // S == T
#define D_DIM 1024
#define NEG_BIG (-1000000000.0f)

typedef __attribute__((ext_vector_type(8)))  float           v8f;
typedef __attribute__((ext_vector_type(8)))  __bf16          bf8;
typedef __attribute__((ext_vector_type(16))) __bf16          bf16v;
typedef __attribute__((ext_vector_type(4)))  unsigned short  us4;
typedef __attribute__((ext_vector_type(4)))  unsigned int    v4u;
typedef __attribute__((ext_vector_type(8)))  int             v8i;
typedef __attribute__((ext_vector_type(4)))  int             v4i;

static __device__ __forceinline__ unsigned short cvt_bf16(float f) {
    unsigned u = __float_as_uint(f);
    u += 0x7FFFu + ((u >> 16) & 1u);            // round-to-nearest-even
    return (unsigned short)(u >> 16);
}

#if HAVE_TDM
// LDS byte offset of a __shared__ object (AS3 pointers are 32-bit offsets)
static __device__ __forceinline__ unsigned lds_off(const void* p) {
    return (unsigned)(size_t)(const __attribute__((address_space(3))) void*)p;
}

// TDM: DMA a 2D bf16 tile [128 rows x 32 elems] (row stride = stride_elems)
// from global memory into LDS at lds_byte_off, inserting 8 halfs (4 dwords)
// of padding after every 32 halfs (16 dwords) -> LDS rows of 40 halfs.
// D# layout per CDNA5 ISA ch.8 (group0 128b, group1 256b; groups2/3 zero=2D).
static __device__ __forceinline__ void tdm_load_tile(
    const unsigned short* gptr, unsigned lds_byte_off, unsigned stride_elems) {
    const unsigned long long ga = (unsigned long long)(size_t)gptr;
    v4u g0;
    g0[0] = 1u;                                     // count=1 (valid), user mode
    g0[1] = lds_byte_off;                           // lds_addr
    g0[2] = (unsigned)(ga & 0xFFFFFFFFu);           // global_addr[31:0]
    g0[3] = (unsigned)((ga >> 32) & 0x01FFFFFFu)    // global_addr[56:32]
          | (2u << 30);                             // type = 2 ("image")
    v8i g1;
    g1[0] = (int)((1u << 16)    // data_size = 2 bytes
                | (1u << 20)    // pad_enable
                | (3u << 22)    // pad_interval: 16 dwords
                | (3u << 25));  // pad_amount: 4 dwords
    const unsigned td0 = 32u, td1 = 128u;           // tensor dims == tile dims
    g1[1] = (int)((td0 & 0xFFFFu) << 16);           // tensor_dim0[15:0]
    g1[2] = (int)(((td0 >> 16) & 0xFFFFu) | ((td1 & 0xFFFFu) << 16));
    g1[3] = (int)(((td1 >> 16) & 0xFFFFu) | (32u << 16));   // tile_dim0 = 32
    g1[4] = (int)(128u);                            // tile_dim1 = 128 rows
    g1[5] = (int)stride_elems;                      // tensor_dim0_stride[31:0]
    g1[6] = 0;
    g1[7] = 0;
    const v4i z4 = {0, 0, 0, 0};
#if TDM_HEADER_TOOLCHAIN
    const v8i z8 = {0, 0, 0, 0, 0, 0, 0, 0};
    __builtin_amdgcn_tensor_load_to_lds(g0, g1, z4, z4, z8, 0);   // 6-arg form
#else
    __builtin_amdgcn_tensor_load_to_lds(g0, g1, z4, z4, 0);       // 5-arg form
#endif
}
#endif  // HAVE_TDM

#if HAVE_ASYNC_LDS
// Per-lane async global->LDS 16-byte copy (ASYNCcnt-tracked).
// Prototype (from round-5 diagnostic): (v4i AS1*, v4i AS3*, imm offset, imm cpol)
static __device__ __forceinline__ void async_b128(
    const unsigned short* g, unsigned short* l) {
    __builtin_amdgcn_global_load_async_to_lds_b128(
        (__attribute__((address_space(1))) v4i*)g,
        (__attribute__((address_space(3))) v4i*)l,
        0, 0);
}
#endif

// ---------------------------------------------------------------------------
// f32 -> bf16 bulk convert (vectorized: global_load_b128 / store_b64)
// ---------------------------------------------------------------------------
__global__ __launch_bounds__(256)
void cvt_f32_bf16_kernel(const float* __restrict__ in,
                         unsigned short* __restrict__ out, int n4) {
    int i = blockIdx.x * 256 + threadIdx.x;
    if (i >= n4) return;
    float4 v = ((const float4*)in)[i];
    us4 o;
    o[0] = cvt_bf16(v.x); o[1] = cvt_bf16(v.y);
    o[2] = cvt_bf16(v.z); o[3] = cvt_bf16(v.w);
    ((us4*)out)[i] = o;
}

// ---------------------------------------------------------------------------
// Tiled bf16 WMMA GEMM, f32 accumulate, double-buffered LDS.
//   BT == false : C[m][n] = sum_k A[m][k] * B[n][k]   (A: MxK, B: NxK, K-contig)
//   BT == true  : C[m][n] = sum_k A[m][k] * B[k][n]   (A: MxK, B: KxN row-major)
//   MASK        : rows m with rowMask[m]==0 are overwritten with NEG_BIG
// Block tile 128x128, K-step 32; 256 threads = 8 waves (2Mx4N), each wave
// computes 64x32 via 4x2 v_wmma_f32_16x16x32_bf16 accumulators.
// Staging: A-tile via TDM (TENSORcnt), NT B-tile via async global->LDS
// (ASYNCcnt), BT B-tile (16-bit transpose) via reg-bounce ds_store.
// ---------------------------------------------------------------------------
template<bool BT, bool MASK>
__global__ __launch_bounds__(256)
void gemm_bf16_kernel(const unsigned short* __restrict__ A,
                      const unsigned short* __restrict__ Bm,
                      float* __restrict__ C,
                      int M, int N, int K,
                      int lda, int ldb, int ldc,
                      const int* __restrict__ rowMask) {
    // +8 half pad per row: rows are 80B apart (16B aligned, conflict-friendly)
    __shared__ unsigned short As[2][128 * 40];
    __shared__ unsigned short Bs[2][128 * 40];

    const int tid  = threadIdx.x;
    const int lane = tid & 31;
    const int wid  = tid >> 5;
    const int wm   = wid >> 2;          // 0..1
    const int wn   = wid & 3;           // 0..3
    const int hi   = lane >> 4;         // 0..1 (K-half select, per ISA layout)
    const int lm   = lane & 15;         // row/col within 16
    const int m0   = blockIdx.y * 128;
    const int n0   = blockIdx.x * 128;

    // staging indices
    const int ar  = tid >> 1;           // 0..127   (row of A/B tile)
    const int ak  = (tid & 1) * 16;     // 0 or 16  (16 halfs = 32B per thread)
    const int bk2 = (tid & 15) * 2;     // 0..30    (BT: K-row pair)
    const int bn8 = (tid >> 4) * 8;     // 0..120   (BT: 8-col slice)

#if HAVE_TDM
    const bool issuer = (wid == 0);     // one wave drives the TDM
    unsigned ldsA[2];
    ldsA[0] = lds_off(&As[0][0]); ldsA[1] = lds_off(&As[1][0]);
#endif

    v8f acc[4][2] = {};

    // ---- prologue: stage K-tile 0 into buffer 0 ----
#if HAVE_TDM
    if (issuer) tdm_load_tile(A + (size_t)m0 * lda, ldsA[0], (unsigned)lda);
#else
    {
        const uint4* g = (const uint4*)(A + (size_t)(m0 + ar) * lda + ak);
        uint4* s = (uint4*)(&As[0][ar * 40 + ak]);
        s[0] = g[0]; s[1] = g[1];
    }
#endif
    if (!BT) {
#if HAVE_ASYNC_LDS
        async_b128(Bm + (size_t)(n0 + ar) * ldb + ak,     &Bs[0][ar * 40 + ak]);
        async_b128(Bm + (size_t)(n0 + ar) * ldb + ak + 8, &Bs[0][ar * 40 + ak + 8]);
#else
        const uint4* g = (const uint4*)(Bm + (size_t)(n0 + ar) * ldb + ak);
        uint4* s = (uint4*)(&Bs[0][ar * 40 + ak]);
        s[0] = g[0]; s[1] = g[1];
#endif
    } else {
        union { uint4 v; unsigned short s[8]; } r0, r1;
        r0.v = *(const uint4*)(Bm + (size_t)bk2 * ldb + n0 + bn8);
        r1.v = *(const uint4*)(Bm + (size_t)(bk2 + 1) * ldb + n0 + bn8);
        #pragma unroll
        for (int j = 0; j < 8; j++) {
            unsigned pk = (unsigned)r0.s[j] | ((unsigned)r1.s[j] << 16);
            *(unsigned*)(&Bs[0][(bn8 + j) * 40 + bk2]) = pk;
        }
    }

    for (int k0 = 0; k0 < K; k0 += 32) {
        const int cur = (k0 >> 5) & 1;
        // my DMA transfers into `cur` must land before anyone reads it
#if HAVE_TDM
        if (issuer) __builtin_amdgcn_s_wait_tensorcnt(0);
#endif
#if HAVE_ASYNC_LDS
        if (!BT) __builtin_amdgcn_s_wait_asynccnt(0);
#endif
        __syncthreads();    // buffer `cur` fully staged; `cur^1` free to fill

        const bool has_next = (k0 + 32) < K;
        const int  kn = k0 + 32;

        // ---- kick off next K-tile fetch (overlaps with the WMMAs) ----
#if HAVE_TDM
        if (has_next && issuer)
            tdm_load_tile(A + (size_t)m0 * lda + kn, ldsA[cur ^ 1], (unsigned)lda);
#else
        uint4 na0, na1;
        if (has_next) {
            const uint4* g = (const uint4*)(A + (size_t)(m0 + ar) * lda + kn + ak);
            na0 = g[0]; na1 = g[1];
        }
#endif
#if !HAVE_ASYNC_LDS
        uint4 nb0, nb1;
#endif
        if (!BT && has_next) {
#if HAVE_ASYNC_LDS
            async_b128(Bm + (size_t)(n0 + ar) * ldb + kn + ak,
                       &Bs[cur ^ 1][ar * 40 + ak]);
            async_b128(Bm + (size_t)(n0 + ar) * ldb + kn + ak + 8,
                       &Bs[cur ^ 1][ar * 40 + ak + 8]);
#else
            const uint4* g = (const uint4*)(Bm + (size_t)(n0 + ar) * ldb + kn + ak);
            nb0 = g[0]; nb1 = g[1];
#endif
        }
        union { uint4 v; unsigned short s[8]; } nr0, nr1;
        if (BT && has_next) {
            nr0.v = *(const uint4*)(Bm + (size_t)(kn + bk2) * ldb + n0 + bn8);
            nr1.v = *(const uint4*)(Bm + (size_t)(kn + bk2 + 1) * ldb + n0 + bn8);
            if (k0 + 64 < K)
                __builtin_prefetch(Bm + (size_t)(k0 + 64 + bk2) * ldb + n0 + bn8, 0, 3);
        }

        // ---- fragments per ISA 16-bit layouts, from buffer `cur` ----
        // A 16x32: lane(lm,hi) holds K = hi*8+{0..7} and 16+hi*8+{0..7}
        bf16v af[4];
        #pragma unroll
        for (int mt = 0; mt < 4; mt++) {
            const unsigned short* p = &As[cur][(wm * 64 + mt * 16 + lm) * 40 + hi * 8];
            bf8 lo = *(const bf8*)p;
            bf8 h2 = *(const bf8*)(p + 16);
            af[mt] = __builtin_shufflevector(lo, h2, 0,1,2,3,4,5,6,7,8,9,10,11,12,13,14,15);
        }
        // B 32x16: lane(lm,hi) holds N=lm, K = hi*8+{0..7} and 16+hi*8+{0..7}
        bf16v bfr[2];
        #pragma unroll
        for (int nt = 0; nt < 2; nt++) {
            const unsigned short* p = &Bs[cur][(wn * 32 + nt * 16 + lm) * 40 + hi * 8];
            bf8 lo = *(const bf8*)p;
            bf8 h2 = *(const bf8*)(p + 16);
            bfr[nt] = __builtin_shufflevector(lo, h2, 0,1,2,3,4,5,6,7,8,9,10,11,12,13,14,15);
        }

        #pragma unroll
        for (int mt = 0; mt < 4; mt++)
            #pragma unroll
            for (int nt = 0; nt < 2; nt++)
                acc[mt][nt] = __builtin_amdgcn_wmma_f32_16x16x32_bf16(
                    false, af[mt], false, bfr[nt], (short)0, acc[mt][nt], false, false);

        // ---- store next tile into the other buffer (manual paths only) ----
#if !HAVE_TDM
        if (has_next) {
            uint4* s = (uint4*)(&As[cur ^ 1][ar * 40 + ak]);
            s[0] = na0; s[1] = na1;
        }
#endif
#if !HAVE_ASYNC_LDS
        if (!BT && has_next) {
            uint4* s = (uint4*)(&Bs[cur ^ 1][ar * 40 + ak]);
            s[0] = nb0; s[1] = nb1;
        }
#endif
        if (BT && has_next) {
            #pragma unroll
            for (int j = 0; j < 8; j++) {
                unsigned pk = (unsigned)nr0.s[j] | ((unsigned)nr1.s[j] << 16);
                *(unsigned*)(&Bs[cur ^ 1][(bn8 + j) * 40 + bk2]) = pk;
            }
        }
    }

    // ---- epilogue: C layout VGPR r -> M = r + 8*hi, N = lm ----
    #pragma unroll
    for (int mt = 0; mt < 4; mt++) {
        const int mbase = m0 + wm * 64 + mt * 16 + hi * 8;
        #pragma unroll
        for (int nt = 0; nt < 2; nt++) {
            const int col = n0 + wn * 32 + nt * 16 + lm;
            #pragma unroll
            for (int r = 0; r < 8; r++) {
                const int row = mbase + r;
                float v = acc[mt][nt][r];
                if (MASK) { if (rowMask[row] == 0) v = NEG_BIG; }
                C[(size_t)row * ldc + col] = v;
            }
        }
    }
}

// ---------------------------------------------------------------------------
// Column softmax over rows of V[R][Ccols] (f32), optional row padding mask
// (rowMask[r]==0 -> value treated as -inf). Writes bf16 output TRANSPOSED:
// OT[c][r], so downstream GEMMs read it K-contiguously.
// ---------------------------------------------------------------------------
template<bool MASK>
__global__ __launch_bounds__(256)
void colsoftmax_T_kernel(const float* __restrict__ V,
                         unsigned short* __restrict__ OT,
                         int R, int Ccols,
                         const int* __restrict__ rowMask) {
    __shared__ float red[4][64];
    const int cx = threadIdx.x & 63;
    const int ty = threadIdx.x >> 6;
    const int c  = blockIdx.x * 64 + cx;

    float m = -__builtin_inff();
    for (int r = ty; r < R; r += 4) {
        float v = V[(size_t)r * Ccols + c];
        if (MASK && rowMask[r] == 0) v = -__builtin_inff();
        m = fmaxf(m, v);
    }
    red[ty][cx] = m;
    __syncthreads();
    m = fmaxf(fmaxf(red[0][cx], red[1][cx]), fmaxf(red[2][cx], red[3][cx]));
    __syncthreads();

    float s = 0.0f;
    for (int r = ty; r < R; r += 4) {
        float v = V[(size_t)r * Ccols + c];
        if (MASK && rowMask[r] == 0) v = -__builtin_inff();
        s += __expf(v - m);
    }
    red[ty][cx] = s;
    __syncthreads();
    s = red[0][cx] + red[1][cx] + red[2][cx] + red[3][cx];
    const float inv = 1.0f / s;

    for (int r = ty; r < R; r += 4) {
        float v = V[(size_t)r * Ccols + c];
        if (MASK && rowMask[r] == 0) v = -__builtin_inff();
        OT[(size_t)c * R + r] = cvt_bf16(__expf(v - m) * inv);
    }
}

// ---------------------------------------------------------------------------
// Pipeline per batch (stream-ordered, workspace ~100 MB):
//   scores = state@x^T (+row mask) -> colsoftmax^T -> P^T
//   w = P^T@state -> colsoftmax^T (row mask) -> attn^T
//   out_b = attn^T@state
// ---------------------------------------------------------------------------
extern "C" void kernel_launch(void* const* d_in, const int* in_sizes, int n_in,
                              void* d_out, int out_size, void* d_ws, size_t ws_size,
                              hipStream_t stream) {
    const float* state = (const float*)d_in[0];
    const float* x     = (const float*)d_in[1];
    const int*   src   = (const int*)d_in[2];   // jax default: int64 -> int32
    float* out = (float*)d_out;

    const size_t nSD = (size_t)B_DIM * S_DIM * D_DIM;

    unsigned short* sbf    = (unsigned short*)d_ws;
    unsigned short* xbf    = sbf + nSD;
    float*          scores = (float*)(xbf + nSD);
    unsigned short* PT     = (unsigned short*)(scores + (size_t)S_DIM * S_DIM);
    float*          w      = (float*)(PT + (size_t)S_DIM * S_DIM);
    unsigned short* attnT  = (unsigned short*)(w + (size_t)S_DIM * D_DIM);

    {
        int n4 = (int)(nSD / 4);
        int blocks = (n4 + 255) / 256;
        cvt_f32_bf16_kernel<<<blocks, 256, 0, stream>>>(state, sbf, n4);
        cvt_f32_bf16_kernel<<<blocks, 256, 0, stream>>>(x, xbf, n4);
    }

    for (int b = 0; b < B_DIM; b++) {
        const unsigned short* sb = sbf + (size_t)b * S_DIM * D_DIM;
        const unsigned short* xb = xbf + (size_t)b * S_DIM * D_DIM;
        const int* srcb = src + (size_t)b * S_DIM;

        gemm_bf16_kernel<false, true>
            <<<dim3(S_DIM / 128, S_DIM / 128), 256, 0, stream>>>(
                sb, xb, scores, S_DIM, S_DIM, D_DIM, D_DIM, D_DIM, S_DIM, srcb);

        colsoftmax_T_kernel<false><<<S_DIM / 64, 256, 0, stream>>>(
            scores, PT, S_DIM, S_DIM, nullptr);

        gemm_bf16_kernel<true, false>
            <<<dim3(D_DIM / 128, S_DIM / 128), 256, 0, stream>>>(
                PT, sb, w, S_DIM, D_DIM, S_DIM, S_DIM, D_DIM, D_DIM, nullptr);

        colsoftmax_T_kernel<true><<<D_DIM / 64, 256, 0, stream>>>(
            w, attnT, S_DIM, D_DIM, srcb);

        gemm_bf16_kernel<true, false>
            <<<dim3(D_DIM / 128, D_DIM / 128), 256, 0, stream>>>(
                attnT, sb, out + (size_t)b * D_DIM * D_DIM,
                D_DIM, D_DIM, S_DIM, S_DIM, D_DIM, D_DIM, nullptr);
    }

    (void)in_sizes; (void)n_in; (void)out_size; (void)ws_size;
}
